// SwinBlock_44186623541517
// MI455X (gfx1250) — compile-verified, matrix operand order checked by reading, not run on hardware
//
#include <hip/hip_runtime.h>
#include <hip/hip_bf16.h>

// SwinBlock (grid attention) for gfx1250 — all GEMMs via v_wmma_f32_16x16x32_f16,
// weight panels staged into LDS with GLOBAL_LOAD_ASYNC_TO_LDS_B128 (ASYNCcnt).
// Workspace layout (needs ~129 MB):
//   [0,32MB)    A_f16  [65536 x 256]  (aliased later by attn-out)
//   [32,64MB)   q_f16  [65536 x 256]  (pre-scaled by D^-0.5)
//   [64,96MB)   k_f16  [65536 x 256]
//   [96,128MB)  vT_f16 [n=256][c=256][l=256]  (transposed for PV B-operand)
//   [128MB..)   f16 copies of in_proj_w (768x256) and out_w (256x256)

typedef __attribute__((ext_vector_type(16))) _Float16 v16h;
typedef __attribute__((ext_vector_type(8)))  _Float16 v8h;
typedef __attribute__((ext_vector_type(8)))  float    v8f;

union FragU { v16h v; v8h h[2]; };

// Load one 16x32 f16 operand fragment (A-layout; B uses the same per-lane
// pattern with "row" = output column, ld = stride between B columns).
// Per CDNA5 ISA 7.12.2: lanes 0-15 row=lane, halves 0-7 K=0..7, 8-15 K=16..23;
// lanes 16-31 row=lane-16, halves 0-7 K=8..15, 8-15 K=24..31.
__device__ __forceinline__ v16h load_frag_f16(const _Float16* base, int ld) {
  const int lane = threadIdx.x & 31;
  const _Float16* p = base + (size_t)(lane & 15) * ld + ((lane & 16) ? 8 : 0);
  FragU f;
  f.h[0] = *(const v8h*)(p);
  f.h[1] = *(const v8h*)(p + 16);
  return f.v;
}

__device__ __forceinline__ v8f wmma_f16(v16h a, v16h b, v8f c) {
  return __builtin_amdgcn_wmma_f32_16x16x32_f16(false, a, false, b, (short)0, c,
                                                false, false);
}

// CDNA5 async copy global -> LDS, 16B per enabled lane (ASYNCcnt-tracked).
// LDS destination offset = low 32 bits of the generic pointer (ISA 10.2 aperture).
__device__ __forceinline__ void async_copy_b128(void* lds_ptr, const void* gptr) {
  unsigned lds_off = (unsigned)(uintptr_t)lds_ptr;
  asm volatile("global_load_async_to_lds_b128 %0, %1, off"
               :: "v"(lds_off), "v"(gptr) : "memory");
}
__device__ __forceinline__ void wait_async0() {
  asm volatile("s_wait_asynccnt 0" ::: "memory");
}

// ---------------- kernel 0: weights fp32 -> f16 ----------------
__global__ void k_cvt_w(const float* __restrict__ wqkv, const float* __restrict__ wout,
                        _Float16* __restrict__ wqkv16, _Float16* __restrict__ wout16) {
  unsigned idx = blockIdx.x * 256u + threadIdx.x;   // < 262144
  if (idx < 196608u) wqkv16[idx] = (_Float16)wqkv[idx];
  else               wout16[idx - 196608u] = (_Float16)wout[idx - 196608u];
}

// ---------------- kernel 1: window rearrange + f16 cast ----------------
// A[m, c] with m = n*256 + l, n = b*64 + h2*8 + w2, l = h1*16 + w1.
__global__ void k_pack(const float* __restrict__ x, _Float16* __restrict__ A) {
  unsigned idx = blockIdx.x * 256u + threadIdx.x;   // < 16777216, x-order (coalesced read)
  unsigned W  = idx & 127u;
  unsigned H  = (idx >> 7) & 127u;
  unsigned ch = (idx >> 14) & 255u;
  unsigned b  = idx >> 22;
  unsigned n  = b * 64u + (H & 7u) * 8u + (W & 7u);
  unsigned l  = (H >> 3) * 16u + (W >> 3);
  unsigned m  = n * 256u + l;
  A[(size_t)m * 256u + ch] = (_Float16)x[idx];
}

// ---------------- kernel 2: fused QKV projection ----------------
// One wave -> 16 rows x 64 cols; the block's 64x256 weight panel is staged
// once into LDS via async-to-LDS. grid = (12 col-blocks, 512 row-blocks) x 256 thr.
__global__ void k_qkv(const _Float16* __restrict__ A, const _Float16* __restrict__ Wf,
                      const float* __restrict__ bias,
                      _Float16* __restrict__ qf, _Float16* __restrict__ kf,
                      _Float16* __restrict__ vT) {
  __shared__ __align__(16) _Float16 bw[64 * 256];   // 32KB weight panel
  const int wave = threadIdx.x >> 5;
  const int lane = threadIdx.x & 31;
  const int m0 = (blockIdx.y * 8 + wave) * 16;
  const int j0 = blockIdx.x * 64;

  // Cooperative async stage: 64*256 halves = 2048 x b128 chunks.
  {
    const _Float16* src = Wf + (size_t)j0 * 256;
    for (int i = threadIdx.x; i < 2048; i += 256)
      async_copy_b128(&bw[i * 8], src + i * 8);
    wait_async0();
  }
  __syncthreads();

  v8f z = {0.f,0.f,0.f,0.f,0.f,0.f,0.f,0.f};
  v8f acc[4] = {z, z, z, z};
#pragma unroll
  for (int kc = 0; kc < 256; kc += 32) {
    v16h a = load_frag_f16(A + (size_t)m0 * 256 + kc, 256);
#pragma unroll
    for (int t = 0; t < 4; ++t) {
      v16h b = load_frag_f16(&bw[(t * 16) * 256 + kc], 256);
      acc[t] = wmma_f16(a, b, acc[t]);
    }
  }
  const int nIdx = lane & 15;
  const int hf   = (lane >> 4) & 1;
  const int region = j0 >> 8;                     // 0=q, 1=k, 2=v
  const float scale = 0.17677669529663687f;       // (D=32)^-0.5 folded into q
#pragma unroll
  for (int t = 0; t < 4; ++t) {
    int j = j0 + t * 16 + nIdx;
    float bj = bias[j];
#pragma unroll
    for (int r = 0; r < 8; ++r) {
      int m = m0 + r + hf * 8;
      float v = acc[t][r] + bj;
      if (region == 0) {
        qf[(size_t)m * 256 + j] = (_Float16)(v * scale);
      } else if (region == 1) {
        kf[(size_t)m * 256 + (j - 256)] = (_Float16)v;
      } else {
        int n = m >> 8, l = m & 255, ch = j - 512;
        vT[((size_t)n * 256 + ch) * 256 + l] = (_Float16)v;   // transposed store
      }
    }
  }
}

// ---------------- kernel 3: attention (QK^T, softmax, PV) ----------------
// One wave -> 16 query rows of one (n,h). grid = 8192 x 128 threads (4 waves).
__global__ void k_attn(const _Float16* __restrict__ q, const _Float16* __restrict__ kmat,
                       const _Float16* __restrict__ vT, _Float16* __restrict__ attn) {
  __shared__ __align__(16) _Float16 plds[4][16 * 264];  // padded stride, 16B-aligned rows
  const int wave = threadIdx.x >> 5;
  const int lane = threadIdx.x & 31;
  const int wid = blockIdx.x * 4 + wave;
  const int rowblk = wid & 15;
  const int nh = wid >> 4;
  const int h = nh & 7;
  const int n = nh >> 3;

  v8f z = {0.f,0.f,0.f,0.f,0.f,0.f,0.f,0.f};
  v16h aq = load_frag_f16(q + ((size_t)(n * 256 + rowblk * 16)) * 256 + h * 32, 256);
  v8f s[16];
#pragma unroll
  for (int j = 0; j < 16; ++j) {            // S = q @ k^T  (K = head dim 32)
    v16h bk = load_frag_f16(kmat + ((size_t)(n * 256 + j * 16)) * 256 + h * 32, 256);
    s[j] = wmma_f16(aq, bk, z);
  }
  // Row softmax: C-layout => lane = column, reg = row (rows split across lane halves).
  float mx[8], inv[8];
#pragma unroll
  for (int r = 0; r < 8; ++r) {
    float m = s[0][r];
#pragma unroll
    for (int j = 1; j < 16; ++j) m = fmaxf(m, s[j][r]);
    m = fmaxf(m, __shfl_xor(m, 1, 32));
    m = fmaxf(m, __shfl_xor(m, 2, 32));
    m = fmaxf(m, __shfl_xor(m, 4, 32));
    m = fmaxf(m, __shfl_xor(m, 8, 32));
    mx[r] = m;
  }
#pragma unroll
  for (int r = 0; r < 8; ++r) {
    float sum = 0.f;
#pragma unroll
    for (int j = 0; j < 16; ++j) { float e = __expf(s[j][r] - mx[r]); s[j][r] = e; sum += e; }
    sum += __shfl_xor(sum, 1, 32);
    sum += __shfl_xor(sum, 2, 32);
    sum += __shfl_xor(sum, 4, 32);
    sum += __shfl_xor(sum, 8, 32);
    inv[r] = 1.0f / sum;
  }
  // Stage P in LDS to convert C-layout -> A-operand layout.
  const int nIdx = lane & 15;
  const int hf   = (lane >> 4) & 1;
#pragma unroll
  for (int j = 0; j < 16; ++j)
#pragma unroll
    for (int r = 0; r < 8; ++r)
      plds[wave][(r + hf * 8) * 264 + j * 16 + nIdx] = (_Float16)s[j][r];
  __syncthreads();

  v8f o0 = z, o1 = z;
#pragma unroll
  for (int kc = 0; kc < 8; ++kc) {          // O = P @ V  (K = 256 keys)
    v16h ap  = load_frag_f16(&plds[wave][kc * 32], 264);
    v16h bv0 = load_frag_f16(vT + ((size_t)(n * 256 + h * 32)) * 256 + kc * 32, 256);
    v16h bv1 = load_frag_f16(vT + ((size_t)(n * 256 + h * 32 + 16)) * 256 + kc * 32, 256);
    o0 = wmma_f16(ap, bv0, o0);
    o1 = wmma_f16(ap, bv1, o1);
  }
#pragma unroll
  for (int r = 0; r < 8; ++r) {
    int m = n * 256 + rowblk * 16 + r + hf * 8;
    attn[(size_t)m * 256 + h * 32 + nIdx]      = (_Float16)(o0[r] * inv[r]);
    attn[(size_t)m * 256 + h * 32 + 16 + nIdx] = (_Float16)(o1[r] * inv[r]);
  }
}

// ---------------- kernel 4: out-proj + LayerNorm + inverse rearrange ----------------
// One wave -> full 16x256 row block so LN stats stay in registers.
// out_w staged into LDS in two 64KB-total phases via async-to-LDS.
__global__ void k_out(const _Float16* __restrict__ attn, const _Float16* __restrict__ Wo,
                      const float* __restrict__ ob, const float* __restrict__ lnw,
                      const float* __restrict__ lnb, float* __restrict__ y) {
  __shared__ __align__(16) _Float16 bw[128 * 256];  // 64KB: half of out_w per phase
  const int wave = threadIdx.x >> 5;
  const int lane = threadIdx.x & 31;
  const int m0 = (blockIdx.x * 8 + wave) * 16;
  v8f z = {0.f,0.f,0.f,0.f,0.f,0.f,0.f,0.f};
  v8f acc[16];
#pragma unroll
  for (int t = 0; t < 16; ++t) acc[t] = z;

  for (int ph = 0; ph < 2; ++ph) {
    __syncthreads();                                // buffer reuse guard
    const _Float16* src = Wo + (size_t)ph * 128 * 256;
    for (int i = threadIdx.x; i < 4096; i += 256)   // 128*256 halves = 4096 b128
      async_copy_b128(&bw[i * 8], src + i * 8);
    wait_async0();
    __syncthreads();
#pragma unroll
    for (int kc = 0; kc < 256; kc += 32) {
      v16h a = load_frag_f16(attn + (size_t)m0 * 256 + kc, 256);
#pragma unroll
      for (int tt = 0; tt < 8; ++tt) {
        v16h b = load_frag_f16(&bw[(tt * 16) * 256 + kc], 256);
        acc[ph * 8 + tt] = wmma_f16(a, b, acc[ph * 8 + tt]);
      }
    }
  }

  const int nIdx = lane & 15;
  const int hf   = (lane >> 4) & 1;
#pragma unroll
  for (int t = 0; t < 16; ++t) {
    float bj = ob[t * 16 + nIdx];
#pragma unroll
    for (int r = 0; r < 8; ++r) acc[t][r] += bj;
  }
  float mean[8], rstd[8];
#pragma unroll
  for (int r = 0; r < 8; ++r) {
    float s = 0.f, q2 = 0.f;
#pragma unroll
    for (int t = 0; t < 16; ++t) { float v = acc[t][r]; s += v; q2 += v * v; }
    s  += __shfl_xor(s, 1, 32);  q2 += __shfl_xor(q2, 1, 32);
    s  += __shfl_xor(s, 2, 32);  q2 += __shfl_xor(q2, 2, 32);
    s  += __shfl_xor(s, 4, 32);  q2 += __shfl_xor(q2, 4, 32);
    s  += __shfl_xor(s, 8, 32);  q2 += __shfl_xor(q2, 8, 32);
    mean[r] = s * (1.0f / 256.0f);
    float var = q2 * (1.0f / 256.0f) - mean[r] * mean[r];
    rstd[r] = rsqrtf(fmaxf(var, 0.0f) + 1e-5f);
  }
#pragma unroll
  for (int t = 0; t < 16; ++t) {
    int j = t * 16 + nIdx;
    float w = lnw[j], b2 = lnb[j];
#pragma unroll
    for (int r = 0; r < 8; ++r) {
      int m = m0 + r + hf * 8;
      int n = m >> 8, l = m & 255;
      int bi = n >> 6, h2 = (n >> 3) & 7, w2 = n & 7;
      int h1 = l >> 4, w1 = l & 15;
      int H = h1 * 8 + h2, W = w1 * 8 + w2;
      size_t off = (((size_t)bi * 256 + j) * 128 + H) * 128 + W;
      y[off] = (acc[t][r] - mean[r]) * rstd[r] * w + b2;
    }
  }
}

extern "C" void kernel_launch(void* const* d_in, const int* in_sizes, int n_in,
                              void* d_out, int out_size, void* d_ws, size_t ws_size,
                              hipStream_t stream) {
  const float* x   = (const float*)d_in[0];
  const float* wq  = (const float*)d_in[1];   // in_proj_w [768,256]
  const float* bq  = (const float*)d_in[2];   // in_proj_b [768]
  const float* wo  = (const float*)d_in[3];   // out_w [256,256]
  const float* bo  = (const float*)d_in[4];   // out_b [256]
  const float* lnw = (const float*)d_in[5];
  const float* lnb = (const float*)d_in[6];

  char* ws = (char*)d_ws;
  const size_t MB = 1024u * 1024u;
  _Float16* A    = (_Float16*)(ws);                 // 32MB, reused as attn-out
  _Float16* qf   = (_Float16*)(ws + 32 * MB);       // 32MB
  _Float16* kf   = (_Float16*)(ws + 64 * MB);       // 32MB
  _Float16* vT   = (_Float16*)(ws + 96 * MB);       // 32MB
  _Float16* w16  = (_Float16*)(ws + 128 * MB);      // 384KB
  _Float16* wo16 = (_Float16*)(ws + 128 * MB + (size_t)768 * 256 * sizeof(_Float16)); // 128KB
  _Float16* attn = A;

  k_cvt_w<<<1024, 256, 0, stream>>>(wq, wo, w16, wo16);
  k_pack <<<65536, 256, 0, stream>>>(x, A);
  k_qkv  <<<dim3(12, 512), 256, 0, stream>>>(A, w16, bq, qf, kf, vT);
  k_attn <<<8192, 128, 0, stream>>>(qf, kf, vT, attn);
  k_out  <<<512, 256, 0, stream>>>(attn, wo16, bo, lnw, lnb, (float*)d_out);
}
